// EISEN_4260607558105
// MI455X (gfx1250) — matrix operand order repaired and checked by reading, not compiled, not run on hardware
//
#include <hip/hip_runtime.h>
#include <hip/hip_bf16.h>

typedef __attribute__((ext_vector_type(16))) _Float16 v16h;
typedef __attribute__((ext_vector_type(8)))  _Float16 v8h;
typedef __attribute__((ext_vector_type(8)))  float    v8f;

#define B_  2
#define C_  32
#define N_  16384           // 128*128
#define S_  1024
#define ROWS_ (B_ * N_)
#define SCALE_ 0.17677669529663689f   // 32^-0.5

// ---------------------------------------------------------------------------
// Kernel 1: transpose [B,C,N] -> per-row packed f16 hi/lo:  row*64 halves,
// halves [0..31] = hi(f16) of the 32 channels, [32..63] = lo residual.
// x = (float)hi + (float)lo to ~2^-22 relative accuracy.
// ---------------------------------------------------------------------------
__global__ __launch_bounds__(256)
void pack_hi_lo(const float* __restrict__ key, const float* __restrict__ qry,
                _Float16* __restrict__ kp, _Float16* __restrict__ qp) {
  int t = blockIdx.x * blockDim.x + threadIdx.x;       // row id = b*N + n
  if (t >= ROWS_) return;
  int b = t / N_, n = t - b * N_;
  const float* ks = key + (size_t)b * C_ * N_ + n;
  const float* qs = qry + (size_t)b * C_ * N_ + n;
  _Float16* kd = kp + (size_t)t * 64;
  _Float16* qd = qp + (size_t)t * 64;
#pragma unroll
  for (int c = 0; c < C_; ++c) {
    float kv = ks[(size_t)c * N_];
    _Float16 kh = (_Float16)kv;
    kd[c]      = kh;
    kd[32 + c] = (_Float16)(kv - (float)kh);
    float qv = qs[(size_t)c * N_];
    _Float16 qh = (_Float16)qv;
    qd[c]      = qh;
    qd[32 + c] = (_Float16)(qv - (float)qh);
  }
}

// ---------------------------------------------------------------------------
// Kernel 2: gather-dot via WMMA.  One wave per query row (b*N+n).
// Per 16 samples: A = 16 gathered key rows (16x32 f16), B = query vector
// replicated into all 16 columns.  hi/lo split -> 3 chained v_wmma for ~fp32
// accuracy.  Column 0 of D (lanes 0 and 16) holds the 16 logits.
//
// A layout (ISA 16-bit A 16x32): lane L serves M = L&15; lanes<16 hold
// K{0..7} (elem 0..7) and K{16..23} (elem 8..15); lanes>=16 hold K{8..15},
// K{24..31}.  B layout: same K order as A (lanes<16 K{0..7},K{16..23};
// lanes>=16 K{8..15},K{24..31}); all columns identical here.
// ---------------------------------------------------------------------------
__global__ __launch_bounds__(256)
void affinity_wmma(const _Float16* __restrict__ kp, const _Float16* __restrict__ qp,
                   const int* __restrict__ sidx, float* __restrict__ logits) {
  const int lane = threadIdx.x & 31;
  const int row  = blockIdx.x * 8 + (threadIdx.x >> 5);   // b*N + n  (exact cover)
  const int m    = lane & 15;
  const int bb   = row / N_;

  // ---- B operand: this row's query, replicated across all 16 columns
  const int kb = (lane < 16) ? 0 : 8;
  const _Float16* q = qp + (size_t)row * 64;
  union { v16h v; v8h h[2]; } bhi, blo;
  bhi.h[0] = *(const v8h*)(q + kb);
  bhi.h[1] = *(const v8h*)(q + kb + 16);
  blo.h[0] = *(const v8h*)(q + 32 + kb);
  blo.h[1] = *(const v8h*)(q + 32 + kb + 16);

  const int k0 = (lane < 16) ? 0 : 8;
  const int* idxrow = sidx + (size_t)row * S_;
  float* out = logits + (size_t)row * S_;

  for (int s0 = 0; s0 < S_; s0 += 16) {
    int idx = idxrow[s0 + m];                      // lane L and L+16 fetch same idx
    if (s0 + 16 < S_)                              // uniform branch: EXEC unchanged
      __builtin_prefetch(idxrow + s0 + 16 + m, 0, 1);

    const _Float16* krow = kp + ((size_t)bb * N_ + (size_t)idx) * 64;
    union { v16h v; v8h h[2]; } ahi, alo;
    ahi.h[0] = *(const v8h*)(krow + k0);           // hi K = k0..k0+7
    ahi.h[1] = *(const v8h*)(krow + k0 + 16);      // hi K = k0+16..k0+23
    alo.h[0] = *(const v8h*)(krow + 32 + k0);      // lo residuals, same K slots
    alo.h[1] = *(const v8h*)(krow + 48 + k0);

    v8f c = {};
    c = __builtin_amdgcn_wmma_f32_16x16x32_f16(false, ahi.v, false, bhi.v,
                                               (short)0, c, false, false);
    c = __builtin_amdgcn_wmma_f32_16x16x32_f16(false, ahi.v, false, blo.v,
                                               (short)0, c, false, false);
    c = __builtin_amdgcn_wmma_f32_16x16x32_f16(false, alo.v, false, bhi.v,
                                               (short)0, c, false, false);

    // D layout: VGPR r / lanes0-15 -> M=r ; lanes16-31 -> M=8+r.  Col 0 only.
    if (lane == 0 || lane == 16) {
      float* o = out + s0 + ((lane >> 4) << 3);
#pragma unroll
      for (int r = 0; r < 8; ++r) o[r] = c[r] * SCALE_;
    }
  }
}

// ---------------------------------------------------------------------------
// Kernel 3: masked-softmax KL loss.  One wave per row; rows with seg==0
// contribute exactly 0 (mask), so skip them.  Logits rows are L2-hot.
// ---------------------------------------------------------------------------
__global__ __launch_bounds__(256)
void loss_kernel(const float* __restrict__ logits, const int* __restrict__ sidx,
                 const int* __restrict__ seg, float* __restrict__ acc) {
  const int lane = threadIdx.x & 31;
  const int row  = blockIdx.x * 8 + (threadIdx.x >> 5);
  const int b    = row / N_;
  const int myseg = seg[row];
  if (myseg == 0) return;                       // uniform per wave

  const float* l   = logits + (size_t)row * S_;
  const int* ir    = sidx + (size_t)row * S_;
  const int* segb  = seg + (size_t)b * N_;

  float mx = -3.4e38f;
  for (int s = lane; s < S_; s += 32) mx = fmaxf(mx, l[s]);
  for (int o = 16; o; o >>= 1) mx = fmaxf(mx, __shfl_xor(mx, o, 32));

  float se = 0.f, cnt = 0.f;
  for (int s = lane; s < S_; s += 32) {
    se  += expf(l[s] - mx);
    cnt += (segb[ir[s]] == myseg) ? 1.f : 0.f;
  }
  for (int o = 16; o; o >>= 1) { se += __shfl_xor(se, o, 32); cnt += __shfl_xor(cnt, o, 32); }

  const float denom = se + 1e-9f;
  const float yt    = 1.f / (cnt + 1e-9f);      // cnt >= 1 (window center = self)
  const float lyt   = logf(fmaxf(yt, 1e-30f));

  float kl = 0.f;
  for (int s = lane; s < S_; s += 32) {
    if (segb[ir[s]] == myseg) {
      float p = expf(l[s] - mx) / denom;
      p = fmaxf(p, 1e-8f);
      kl += yt * (lyt - logf(p));
    }
  }
  for (int o = 16; o; o >>= 1) kl += __shfl_xor(kl, o, 32);

  if (lane == 0) { atomicAdd(&acc[0], kl); atomicAdd(&acc[1], 1.f); }
}

__global__ void finalize_loss(const float* __restrict__ acc, float* __restrict__ out) {
  out[(size_t)ROWS_ * S_] = acc[0] / (acc[1] + 1e-9f);
}

// ---------------------------------------------------------------------------
extern "C" void kernel_launch(void* const* d_in, const int* in_sizes, int n_in,
                              void* d_out, int out_size, void* d_ws, size_t ws_size,
                              hipStream_t stream) {
  const float* key  = (const float*)d_in[0];   // [B,C,H,W] f32
  const float* qry  = (const float*)d_in[1];   // [B,C,H,W] f32
  const int*   segt = (const int*)d_in[2];     // [B,1,H,W] i32
  const int*   sidx = (const int*)d_in[3];     // [B,N,S]   i32
  float* out = (float*)d_out;                  // logits [B,N,S] f32, then loss

  float*    acc = (float*)d_ws;                                 // 2 accumulators
  _Float16* kp  = (_Float16*)((char*)d_ws + 1024);              // [ROWS][64] halves
  _Float16* qp  = kp + (size_t)ROWS_ * 64;                      // [ROWS][64] halves

  hipMemsetAsync(d_ws, 0, 16, stream);

  pack_hi_lo   <<<(ROWS_ + 255) / 256, 256, 0, stream>>>(key, qry, kp, qp);
  affinity_wmma<<<ROWS_ / 8,           256, 0, stream>>>(kp, qp, sidx, out);
  loss_kernel  <<<ROWS_ / 8,           256, 0, stream>>>(out, sidx, segt, acc);
  finalize_loss<<<1, 1, 0, stream>>>(acc, out);
}
